// HybridQuantumNet_23519240913595
// MI455X (gfx1250) — matrix operand order repaired and checked by reading, not compiled, not run on hardware
//
#include <hip/hip_runtime.h>
#include <cstdint>
#include <cstddef>

// ---------------------------------------------------------------------------
// CDNA5 (gfx1250) hybrid quantum-CNN pipeline.
//  conv1 (scalar, K=9 too small for WMMA)  -> f16 (B,14,14,8)
//  conv2 + ReLU + fused 2x2 maxpool (WMMA 16x16x32 f16, K=72->96) -> f16 (B,49,16)
//  conv3 + ReLU (WMMA 16x16x32 f16, K=144->160, TDM-staged LDS)   -> f16 (B,49,16)
//  adaptive-pool+FC folded into G, tanh*pi, 4-qubit statevector, classifier
// ---------------------------------------------------------------------------

typedef __attribute__((ext_vector_type(16))) _Float16 v16h;
typedef __attribute__((ext_vector_type(8)))  float    v8f;
typedef __attribute__((ext_vector_type(4)))  uint32_t v4u;
typedef __attribute__((ext_vector_type(8)))  uint32_t v8u;

union V16 { v16h h; uint4 q[2]; };

// AdaptiveAvgPool2d(7 -> 4) bin weights (torch boundaries)
__constant__ float MPOOL[4][7] = {
  {0.5f, 0.5f, 0.f, 0.f, 0.f, 0.f, 0.f},
  {0.f, 1.f/3.f, 1.f/3.f, 1.f/3.f, 0.f, 0.f, 0.f},
  {0.f, 0.f, 0.f, 1.f/3.f, 1.f/3.f, 1.f/3.f, 0.f},
  {0.f, 0.f, 0.f, 0.f, 0.f, 0.5f, 0.5f}};

// ---------------------------------------------------------------------------
// Tensor Data Mover: 1-D contiguous global->LDS DMA, nbytes must be a
// multiple of 8. Descriptor per CDNA5 ISA:
//  group0: [1:0]count=1 | lds_addr | global_addr[56:0] | type=2
//  group1: data_size=8B, tensor_dim0 = tile_dim0 = nbytes/8, tile_dim1 unused,
//          tensor_dim0_stride = nbytes/8  (single row)
// Issue from ONE wave only; completion via s_wait_tensorcnt 0.
// ---------------------------------------------------------------------------
__device__ __forceinline__ void tdm_load_1d(uint32_t lds_off, const void* gptr,
                                            uint32_t n8 /* # of 8-byte units */)
{
  const uint64_t ga = (uint64_t)(uintptr_t)gptr;
  v4u g0;
  g0[0] = 0x1u;                                    // count=1, user mode
  g0[1] = lds_off;                                 // LDS byte address
  g0[2] = (uint32_t)ga;                            // global_addr[31:0]
  g0[3] = (uint32_t)(ga >> 32) | (2u << 30);       // global_addr[56:32] | type=2
  v8u g1;
  g1[0] = 3u << 16;                                // data_size = 8 bytes
  g1[1] = (n8 & 0xffffu) << 16;                    // tensor_dim0[15:0]
  g1[2] = (n8 >> 16) | (1u << 16);                 // tensor_dim0[31:16] | tensor_dim1=1
  g1[3] = (n8 & 0xffffu) << 16;                    // tile_dim0 = n8
  g1[4] = 0u;                                      // tile_dim1=0 (1-D), tile_dim2=0
  g1[5] = n8;                                      // tensor_dim0_stride[31:0]
  g1[6] = 0u;
  g1[7] = 0u;
  asm volatile("tensor_load_to_lds %0, %1" :: "s"(g0), "s"(g1) : "memory");
}

// ---------------------------------------------------------------------------
// Prep: transpose/pad conv2+conv3 weights to f16 [N=16][K] layouts matching the
// WMMA B-fragment gather; fold pool matrices into FC -> G[pix][c][o];
// precompute cos/sin of qweights/2.
// ---------------------------------------------------------------------------
__global__ __launch_bounds__(256) void prep_kernel(
    const float* __restrict__ w2, const float* __restrict__ w3,
    const float* __restrict__ fcw, const float* __restrict__ qw,
    _Float16* __restrict__ W2T, _Float16* __restrict__ W3T,
    float* __restrict__ G, float* __restrict__ qtrig)
{
  const int tid = threadIdx.x;
  // conv2: K = (ky*3+kx)*8 + c  (72 real, pad to 96)
  for (int k = tid; k < 16*96; k += 256) {
    int o = k / 96, kk = k % 96;
    W2T[k] = (kk < 72) ? (_Float16)w2[o*72 + (kk & 7)*9 + (kk >> 3)] : (_Float16)0.f;
  }
  // conv3: K = (ky*3+kx)*16 + c (144 real, pad to 160)
  for (int k = tid; k < 16*160; k += 256) {
    int o = k / 160, kk = k % 160;
    W3T[k] = (kk < 144) ? (_Float16)w3[o*144 + (kk & 15)*9 + (kk >> 4)] : (_Float16)0.f;
  }
  // G[(pix*16 + c)*4 + o] = sum_{p,q} M[p][i] M[q][j] fcw[o][c*16 + p*4 + q]
  for (int it = tid; it < 3136; it += 256) {
    int o = it & 3, c = (it >> 2) & 15, pix = it >> 6;
    int i = pix / 7, j = pix % 7;
    float s = 0.f;
    for (int p = 0; p < 4; ++p) {
      float wr = MPOOL[p][i];
      if (wr != 0.f)
        for (int qq = 0; qq < 4; ++qq) {
          float wc = MPOOL[qq][j];
          if (wc != 0.f) s += wr * wc * fcw[o*256 + c*16 + p*4 + qq];
        }
    }
    G[it] = s;
  }
  if (tid < 32) {
    float th = qw[tid] * 0.5f;
    qtrig[tid*2]   = cosf(th);
    qtrig[tid*2+1] = sinf(th);
  }
}

// ---------------------------------------------------------------------------
// conv1: (B,1,28,28) f32, stride 2, pad 1, ReLU -> channel-last f16 (B,14,14,8)
// ---------------------------------------------------------------------------
__global__ __launch_bounds__(256) void conv1_kernel(
    const float* __restrict__ x, const float* __restrict__ w,
    const float* __restrict__ bias, _Float16* __restrict__ out, int B)
{
  const int idx = blockIdx.x * 256 + threadIdx.x;
  if (idx >= B * 196) return;
  const int b = idx / 196, p = idx % 196;
  const int oy = p / 14, ox = p % 14;
  const float* xin = x + (size_t)b * 784;
  float acc[8];
#pragma unroll
  for (int o = 0; o < 8; ++o) acc[o] = bias[o];
#pragma unroll
  for (int ky = 0; ky < 3; ++ky) {
    const int iy = oy*2 - 1 + ky;
    if ((unsigned)iy >= 28u) continue;
#pragma unroll
    for (int kx = 0; kx < 3; ++kx) {
      const int ix = ox*2 - 1 + kx;
      if ((unsigned)ix >= 28u) continue;
      const float v = xin[iy*28 + ix];
#pragma unroll
      for (int o = 0; o < 8; ++o) acc[o] = fmaf(v, w[o*9 + ky*3 + kx], acc[o]);
    }
  }
  union { uint4 q; _Float16 h[8]; } st;
#pragma unroll
  for (int o = 0; o < 8; ++o) st.h[o] = (_Float16)fmaxf(acc[o], 0.f);
  *(uint4*)(out + (size_t)idx * 8) = st.q;
}

// ---------------------------------------------------------------------------
// conv2 + ReLU + fused 2x2 maxpool. One wave = one 16-pixel tile (2 rows x 8
// cols of the 14x14 grid) x 16 output channels. Pool folded via intra-lane
// max over VGPR pairs + shfl_xor(16) across lane halves (C-matrix layout:
// lanes 0-15 hold M=r, lanes 16-31 hold M=r+8).
// grid = (B, 4), block = 128 (4 waves); tiles t = by*4+wave, active t<14.
// ---------------------------------------------------------------------------
__global__ __launch_bounds__(128) void conv2_kernel(
    const _Float16* __restrict__ A1, const _Float16* __restrict__ W2T,
    const float* __restrict__ bias, _Float16* __restrict__ P)
{
  __shared__ __align__(16) _Float16 Wlds[16*96];
  __shared__ __align__(16) _Float16 Alds[4][16*96];
  const int tid = threadIdx.x, lane = tid & 31, wv = tid >> 5;
  const size_t b = blockIdx.x;

  for (int i = tid; i < 192; i += 128)               // 16*96 halves = 192 uint4
    ((uint4*)Wlds)[i] = ((const uint4*)W2T)[i];

  const int t = blockIdx.y * 4 + wv;                 // wave-uniform
  const bool active = t < 14;
  const int rb = t >> 1, cb = t & 1;

  if (active) {
    const uint4 z4 = make_uint4(0u,0u,0u,0u);
    for (int i = lane; i < 192; i += 32) ((uint4*)Alds[wv])[i] = z4;
    for (int it = lane; it < 144; it += 32) {        // 16 rows x 9 taps
      const int m = it / 9, n = it % 9;
      const int py = 2*rb + (m >> 3);
      const int px = cb*8 + (m & 7);
      const int iy = py + n/3 - 1;
      const int ix = px + n%3 - 1;
      if ((unsigned)iy < 14u && (unsigned)ix < 14u && px < 14)
        *(uint4*)(&Alds[wv][m*96 + n*8]) =
            *(const uint4*)(A1 + (b*196 + (size_t)(iy*14 + ix)) * 8);
    }
  }
  __syncthreads();

  if (active) {
    v8f acc = {0.f,0.f,0.f,0.f,0.f,0.f,0.f,0.f};
    const int m   = lane & 15;                       // A row / B(N) column
    const int hiA = (lane < 16) ? 0 : 8;
    const int hiB = (lane < 16) ? 0 : 8;
    const _Float16* arow = &Alds[wv][m*96];
    const _Float16* brow = &Wlds[m*96];
#pragma unroll
    for (int s = 0; s < 3; ++s) {
      V16 af, bf;
      af.q[0] = *(const uint4*)(arow + s*32 + hiA);
      af.q[1] = *(const uint4*)(arow + s*32 + hiA + 16);
      bf.q[0] = *(const uint4*)(brow + s*32 + hiB);
      bf.q[1] = *(const uint4*)(brow + s*32 + hiB + 16);
      acc = __builtin_amdgcn_wmma_f32_16x16x32_f16(false, af.h, false, bf.h,
                                                   (short)0, acc, false, false);
    }
    const float bn = bias[m];
    float v[8];
#pragma unroll
    for (int r = 0; r < 8; ++r) v[r] = fmaxf(acc[r] + bn, 0.f);   // bias + ReLU
#pragma unroll
    for (int j = 0; j < 4; ++j) {                    // 2x2 maxpool
      float rm = fmaxf(v[2*j], v[2*j+1]);
      rm = fmaxf(rm, __shfl_xor(rm, 16, 32));
      const int col = cb*4 + j;
      if (lane < 16 && col < 7)
        P[(b*49 + (size_t)(rb*7 + col))*16 + m] = (_Float16)rm;
    }
  }
}

// ---------------------------------------------------------------------------
// conv3 + ReLU on the pooled (7,7,16) grid. One block per sample, 4 waves,
// wave w covers pixels [w*16, w*16+16) of the 49 (last tile padded).
// LDS staging of the input tile (1568 B) and weights (5120 B) is done by the
// Tensor Data Mover (wave 0 issues two tensor_load_to_lds, waits TENSORcnt),
// overlapping with the A-tile zeroing done by all waves.
// ---------------------------------------------------------------------------
__global__ __launch_bounds__(128) void conv3_kernel(
    const _Float16* __restrict__ P, const _Float16* __restrict__ W3T,
    const float* __restrict__ bias, _Float16* __restrict__ O3)
{
  __shared__ __align__(16) _Float16 Plds[784];       // 49*16
  __shared__ __align__(16) _Float16 Wlds[16*160];
  __shared__ __align__(16) _Float16 Alds[4][16*160];
  const int tid = threadIdx.x, lane = tid & 31, wv = tid >> 5;
  const size_t b = blockIdx.x;

  if (wv == 0) {                                     // one wave drives the TDM
    tdm_load_1d((uint32_t)(uintptr_t)&Plds[0], P + b*784, 784*2/8);
    tdm_load_1d((uint32_t)(uintptr_t)&Wlds[0], W3T,       16*160*2/8);
  }
  {
    const uint4 z4 = make_uint4(0u,0u,0u,0u);
    for (int i = lane; i < 320; i += 32) ((uint4*)Alds[wv])[i] = z4;
  }
  if (wv == 0) __builtin_amdgcn_s_wait_tensorcnt(0); // DMA done before barrier
  __syncthreads();

  for (int it = lane; it < 144; it += 32) {
    const int r = it / 9, n = it % 9;
    const int p = wv*16 + r;
    if (p < 49) {
      const int iy = p/7 + n/3 - 1;
      const int ix = p%7 + n%3 - 1;
      if ((unsigned)iy < 7u && (unsigned)ix < 7u) {
        const uint4* src = (const uint4*)(&Plds[(iy*7 + ix)*16]);
        uint4* dst = (uint4*)(&Alds[wv][r*160 + n*16]);
        dst[0] = src[0];
        dst[1] = src[1];
      }
    }
  }
  __syncthreads();

  v8f acc = {0.f,0.f,0.f,0.f,0.f,0.f,0.f,0.f};
  const int m   = lane & 15;
  const int hiA = (lane < 16) ? 0 : 8;
  const int hiB = (lane < 16) ? 0 : 8;
  const _Float16* arow = &Alds[wv][m*160];
  const _Float16* brow = &Wlds[m*160];
#pragma unroll
  for (int s = 0; s < 5; ++s) {
    V16 af, bf;
    af.q[0] = *(const uint4*)(arow + s*32 + hiA);
    af.q[1] = *(const uint4*)(arow + s*32 + hiA + 16);
    bf.q[0] = *(const uint4*)(brow + s*32 + hiB);
    bf.q[1] = *(const uint4*)(brow + s*32 + hiB + 16);
    acc = __builtin_amdgcn_wmma_f32_16x16x32_f16(false, af.h, false, bf.h,
                                                 (short)0, acc, false, false);
  }
  const float bn = bias[m];
#pragma unroll
  for (int r = 0; r < 8; ++r) {
    const int p = wv*16 + r + ((lane < 16) ? 0 : 8);
    if (p < 49)
      O3[(b*49 + (size_t)p)*16 + m] = (_Float16)fmaxf(acc[r] + bn, 0.f);
  }
}

// ---------------------------------------------------------------------------
// Head: pooled-FC via G (LDS broadcast), tanh*pi, exact 4-qubit statevector
// (wire 0 = MSB, mask 8>>w), PauliZ expvals, 4->32 ReLU -> 10.
// One thread per sample.
// ---------------------------------------------------------------------------
__global__ __launch_bounds__(256) void head_kernel(
    const _Float16* __restrict__ O3, const float* __restrict__ G,
    const float* __restrict__ qtrig, const float* __restrict__ fcb,
    const float* __restrict__ c1w, const float* __restrict__ c1b,
    const float* __restrict__ c2w, const float* __restrict__ c2b,
    float* __restrict__ out, int B)
{
  __shared__ float Gl[3136];
  __shared__ float Ql[64];
  for (int i = threadIdx.x; i < 3136; i += 256) Gl[i] = G[i];
  if (threadIdx.x < 64) Ql[threadIdx.x] = qtrig[threadIdx.x];
  __syncthreads();

  const int b = blockIdx.x * 256 + threadIdx.x;
  if (b >= B) return;

  float q0 = fcb[0], q1 = fcb[1], q2 = fcb[2], q3 = fcb[3];
  const uint4* hb = (const uint4*)(O3 + (size_t)b * 784);
  for (int pix = 0; pix < 49; ++pix) {
    __builtin_prefetch(hb + (pix + 8)*2, 0, 1);      // global_prefetch_b8
    union { uint4 u[2]; _Float16 h[16]; } d;
    d.u[0] = hb[pix*2];
    d.u[1] = hb[pix*2 + 1];
    const float* g = &Gl[pix*64];
#pragma unroll
    for (int c = 0; c < 16; ++c) {
      const float f = (float)d.h[c];
      q0 = fmaf(f, g[c*4+0], q0);
      q1 = fmaf(f, g[c*4+1], q1);
      q2 = fmaf(f, g[c*4+2], q2);
      q3 = fmaf(f, g[c*4+3], q3);
    }
  }
  const float PI = 3.14159265358979323846f;
  float ang[4] = { tanhf(q0)*PI, tanhf(q1)*PI, tanhf(q2)*PI, tanhf(q3)*PI };

  float ar[16], ai[16];
#pragma unroll
  for (int i = 0; i < 16; ++i) { ar[i] = 0.f; ai[i] = 0.f; }
  ar[0] = 1.f;

  // angle encoding: RY(ang[w]) on wire w
#pragma unroll
  for (int w = 0; w < 4; ++w) {
    const float th = ang[w] * 0.5f;
    const float cc = cosf(th), ss = sinf(th);
    const int mk = 8 >> w;
#pragma unroll
    for (int i = 0; i < 16; ++i)
      if (!(i & mk)) {
        const int j = i | mk;
        const float r0 = ar[i], i0 = ai[i], r1 = ar[j], i1 = ai[j];
        ar[i] = cc*r0 - ss*r1;  ai[i] = cc*i0 - ss*i1;
        ar[j] = ss*r0 + cc*r1;  ai[j] = ss*i0 + cc*i1;
      }
  }
  // variational layers
#pragma unroll
  for (int l = 0; l < 4; ++l) {
#pragma unroll
    for (int w = 0; w < 4; ++w) {
      const int base = ((l*4 + w)*2)*2;
      const float cy = Ql[base+0], sy = Ql[base+1];
      const float cz = Ql[base+2], sz = Ql[base+3];
      const int mk = 8 >> w;
#pragma unroll
      for (int i = 0; i < 16; ++i)                   // RY
        if (!(i & mk)) {
          const int j = i | mk;
          const float r0 = ar[i], i0 = ai[i], r1 = ar[j], i1 = ai[j];
          ar[i] = cy*r0 - sy*r1;  ai[i] = cy*i0 - sy*i1;
          ar[j] = sy*r0 + cy*r1;  ai[j] = sy*i0 + cy*i1;
        }
#pragma unroll
      for (int i = 0; i < 16; ++i) {                 // RZ: e^{-+ i th/2}
        const float xx = ar[i], yy = ai[i];
        if (i & mk) { ar[i] = xx*cz - yy*sz; ai[i] = xx*sz + yy*cz; }
        else        { ar[i] = xx*cz + yy*sz; ai[i] = yy*cz - xx*sz; }
      }
    }
#pragma unroll
    for (int e = 0; e < 4; ++e) {                    // CNOT ring
      const int mc = 8 >> e, mt = 8 >> ((e+1) & 3);
#pragma unroll
      for (int i = 0; i < 16; ++i)
        if ((i & mc) && !(i & mt)) {
          const int j = i | mt;
          const float tr = ar[i], ti = ai[i];
          ar[i] = ar[j]; ai[i] = ai[j];
          ar[j] = tr;    ai[j] = ti;
        }
    }
  }
  float pr[16];
#pragma unroll
  for (int i = 0; i < 16; ++i) pr[i] = ar[i]*ar[i] + ai[i]*ai[i];
  float z[4];
#pragma unroll
  for (int w = 0; w < 4; ++w) {
    const int mk = 8 >> w;
    float s = 0.f;
#pragma unroll
    for (int i = 0; i < 16; ++i) s += (i & mk) ? -pr[i] : pr[i];
    z[w] = s;
  }
  float o[10];
#pragma unroll
  for (int j = 0; j < 10; ++j) o[j] = c2b[j];
#pragma unroll 4
  for (int k = 0; k < 32; ++k) {
    float s = c1b[k];
#pragma unroll
    for (int w = 0; w < 4; ++w) s = fmaf(z[w], c1w[k*4 + w], s);
    s = fmaxf(s, 0.f);
#pragma unroll
    for (int j = 0; j < 10; ++j) o[j] = fmaf(s, c2w[j*32 + k], o[j]);
  }
#pragma unroll
  for (int j = 0; j < 10; ++j) out[(size_t)b*10 + j] = o[j];
}

// ---------------------------------------------------------------------------
extern "C" void kernel_launch(void* const* d_in, const int* in_sizes, int n_in,
                              void* d_out, int out_size, void* d_ws, size_t ws_size,
                              hipStream_t stream)
{
  (void)n_in; (void)out_size; (void)ws_size;
  const float* x   = (const float*)d_in[0];
  const float* w1  = (const float*)d_in[1];
  const float* b1  = (const float*)d_in[2];
  const float* w2  = (const float*)d_in[3];
  const float* b2  = (const float*)d_in[4];
  const float* w3  = (const float*)d_in[5];
  const float* b3  = (const float*)d_in[6];
  const float* fcw = (const float*)d_in[7];
  const float* fcb = (const float*)d_in[8];
  const float* qw  = (const float*)d_in[9];
  const float* c1w = (const float*)d_in[10];
  const float* c1b = (const float*)d_in[11];
  const float* c2w = (const float*)d_in[12];
  const float* c2b = (const float*)d_in[13];
  const int B = in_sizes[0] / 784;

  // workspace layout (peak ~147 MB): weights | A1 (B*14*14*8 f16) | P (B*49*16 f16)
  // O3 aliases A1 (dead after conv2).
  char* ws = (char*)d_ws;
  _Float16* W2T  = (_Float16*)(ws + 0);           //  3 KB
  _Float16* W3T  = (_Float16*)(ws + 4096);        //  5 KB
  float*    G    = (float*)(ws + 12288);          // 12.25 KB
  float*    qtrg = (float*)(ws + 28672);          // 256 B
  _Float16* A1   = (_Float16*)(ws + 65536);       // B*3136 bytes
  _Float16* P    = (_Float16*)(ws + 65536 + (size_t)B * 3136);  // B*1568 bytes
  _Float16* O3   = A1;

  prep_kernel <<<1, 256, 0, stream>>>(w2, w3, fcw, qw, W2T, W3T, G, qtrg);
  conv1_kernel<<<(B*196 + 255)/256, 256, 0, stream>>>(x, w1, b1, A1, B);
  conv2_kernel<<<dim3((unsigned)B, 4), 128, 0, stream>>>(A1, W2T, b2, P);
  conv3_kernel<<<B, 128, 0, stream>>>(P, W3T, b3, O3);
  head_kernel <<<(B + 255)/256, 256, 0, stream>>>(O3, G, qtrg, fcb,
                                                  c1w, c1b, c2w, c2b,
                                                  (float*)d_out, B);
}
